// Encoder_81509889343954
// MI455X (gfx1250) — matrix operand order repaired and checked by reading, not compiled, not run on hardware
//
#include <hip/hip_runtime.h>
#include <hip/hip_bf16.h>

typedef __attribute__((ext_vector_type(8)))  float  v8f;
typedef __attribute__((ext_vector_type(16))) __bf16 v16bf;

#define NN      100000
#define NE      1600000
#define INDIM   256
#define HIDDIM  64
#define LATDIM  32
#define EPSV    1e-5f

__device__ __forceinline__ void atomAddF(float* p, float v) {
    __hip_atomic_fetch_add(p, v, __ATOMIC_RELAXED, __HIP_MEMORY_SCOPE_AGENT);
}

// ---------------- degree / norm precompute ----------------

__global__ void deg_kernel(const int* __restrict__ ei, const float* __restrict__ w,
                           float* __restrict__ deg, int E) {
    int e = blockIdx.x * blockDim.x + threadIdx.x;
    if (e < E) atomAddF(&deg[ei[E + e]], w[e]);   // col = ei[1][e]
}

__global__ void dinv_kernel(float* __restrict__ deg, int n) {
    int i = blockIdx.x * blockDim.x + threadIdx.x;
    if (i < n) {
        float d = deg[i] + 1.0f;                  // +1 self-loop weight
        deg[i] = rsqrtf(d);                       // d > 0 always
    }
}

__global__ void norm_kernel(const int* __restrict__ ei, const float* __restrict__ w,
                            const float* __restrict__ dinv, float* __restrict__ norm, int E) {
    int e = blockIdx.x * blockDim.x + threadIdx.x;
    if (e < E) norm[e] = dinv[ei[e]] * w[e] * dinv[ei[E + e]];
}

// ---------------- weight packing (f32 -> bf16 row-major) ----------------

__global__ void pack_w1wr(const float* __restrict__ W1, const float* __restrict__ Wr,
                          __bf16* __restrict__ Wc) {
    int t = blockIdx.x * blockDim.x + threadIdx.x;
    if (t >= INDIM * 96) return;
    int k = t / 96, n = t % 96;
    float v = (n < HIDDIM) ? W1[k * HIDDIM + n] : Wr[k * LATDIM + (n - HIDDIM)];
    Wc[t] = (__bf16)v;
}

__global__ void pack_w2(const float* __restrict__ W2, __bf16* __restrict__ Wp) {
    int t = blockIdx.x * blockDim.x + threadIdx.x;
    if (t < HIDDIM * LATDIM) Wp[t] = (__bf16)W2[t];
}

// ---------------- B reorder: row-major bf16 -> WMMA-fragment-major ----------------
// dst[(((t*S)+s)*32 + lane)*16 + j] = src[(s*32 + kbB(lane) + (j&7) + (j>>3)*16)*ldb + t*16 + (lane&15)]
// so each lane's 16 B values for (tile t, K-step s) are 32 contiguous bytes.

__global__ void reorder_b_kernel(const __bf16* __restrict__ src, int ldb, int S, int NT,
                                 __bf16* __restrict__ dst) {
    int tid = blockIdx.x * blockDim.x + threadIdx.x;
    int total = NT * S * 32 * 16;
    if (tid >= total) return;
    int j    = tid & 15;
    int lane = (tid >> 4) & 31;
    int s    = (tid >> 9) % S;
    int t    = tid / (S * 512);
    int kbB  = (lane >> 4) * 8;
    int n    = t * 16 + (lane & 15);
    int k    = s * 32 + kbB + (j & 7) + (j >> 3) * 16;
    dst[tid] = src[k * ldb + n];
}

// ---------------- WMMA GEMM: C[N x NT*16] = A[N x K](f32) * Bpack(bf16 frag-major) ---------
// One wave per 16-row strip; NT accumulators so A is streamed exactly once.
// A fragment (ISA 7.12.2, 16-bit A 16x32): lane m = lane&15; lanes 0-15 hold K 0-7 & 16-23,
// lanes 16-31 hold K 8-15 & 24-31 -> 4x float4 loads per K-step.
// C/D: vgpr r -> row r (+8 for upper lanes), col = lane&15.

template <int NT>
__global__ void wmma_gemm_bf16(const float* __restrict__ A, int lda,
                               const __bf16* __restrict__ Bpack,
                               float* __restrict__ C, int ldc, int K, int nRowTiles) {
    const int lane    = threadIdx.x & 31;
    const int rowTile = blockIdx.x * (blockDim.x >> 5) + (threadIdx.x >> 5);
    if (rowTile >= nRowTiles) return;           // wave-uniform: EXEC stays all-1s for WMMA
    const int rowbase = rowTile * 16;
    const int m   = lane & 15;
    const int kbA = (lane >> 4) * 8;
    const int S   = K >> 5;

    const float* Ap = A + (size_t)(rowbase + m) * lda;

    v8f acc[NT] = {};
    for (int k0 = 0; k0 < K; k0 += 32) {
        const float4 f0 = *(const float4*)(Ap + k0 + kbA);
        const float4 f1 = *(const float4*)(Ap + k0 + kbA + 4);
        const float4 f2 = *(const float4*)(Ap + k0 + 16 + kbA);
        const float4 f3 = *(const float4*)(Ap + k0 + 16 + kbA + 4);
        v16bf a;
        a[0]  = (__bf16)f0.x; a[1]  = (__bf16)f0.y; a[2]  = (__bf16)f0.z; a[3]  = (__bf16)f0.w;
        a[4]  = (__bf16)f1.x; a[5]  = (__bf16)f1.y; a[6]  = (__bf16)f1.z; a[7]  = (__bf16)f1.w;
        a[8]  = (__bf16)f2.x; a[9]  = (__bf16)f2.y; a[10] = (__bf16)f2.z; a[11] = (__bf16)f2.w;
        a[12] = (__bf16)f3.x; a[13] = (__bf16)f3.y; a[14] = (__bf16)f3.z; a[15] = (__bf16)f3.w;

        const int s = k0 >> 5;
#pragma unroll
        for (int t = 0; t < NT; ++t) {
            const v16bf b = *(const v16bf*)(Bpack + ((size_t)(t * S + s) * 32 + lane) * 16);
            acc[t] = __builtin_amdgcn_wmma_f32_16x16x32_bf16(
                         /*neg_a=*/false, a, /*neg_b=*/false, b,
                         /*c_mod=*/(short)0, acc[t], /*reuse_a=*/false, /*reuse_b=*/false);
        }
    }

    const int mo = (lane >> 4) * 8;
#pragma unroll
    for (int t = 0; t < NT; ++t) {
        float* Cp = C + (size_t)rowbase * ldc + t * 16 + (lane & 15);
#pragma unroll
        for (int r = 0; r < 8; ++r)
            Cp[(size_t)(r + mo) * ldc] = acc[t][r];
    }
}

// ---------------- edge scatter-add: agg[col] += h[row] * norm ----------------
// Items E..E+n-1 are self loops with weight dinv^2. F/2 threads per item (float2 each),
// so one wave covers 64 consecutive floats of one message -> coalesced.

__global__ void scatter_kernel(const float* __restrict__ h, int hld,
                               const int* __restrict__ ei,
                               const float* __restrict__ norm,
                               const float* __restrict__ dinv,
                               float* __restrict__ agg, int F, int E, int n) {
    const int half = F >> 1;
    long long idx   = (long long)blockIdx.x * blockDim.x + threadIdx.x;
    long long total = (long long)(E + n) * half;
    if (idx >= total) return;
    int item = (int)(idx / half);
    int f    = (int)(idx % half) * 2;
    int r, c; float w;
    if (item < E) { r = ei[item]; c = ei[E + item]; w = norm[item]; }
    else          { r = c = item - E; float d = dinv[r]; w = d * d; }
    const float* hp = h + (size_t)r * hld + f;
    float v0 = hp[0] * w, v1 = hp[1] * w;
    float* ap = agg + (size_t)c * F + f;
    atomAddF(ap,     v0);
    atomAddF(ap + 1, v1);
}

// ---------------- batchnorm: per-channel sum / sumsq (bias folded in) ----------------

__global__ void bn_stats_kernel(const float* __restrict__ x, const float* __restrict__ bias,
                                float* __restrict__ gsum, float* __restrict__ gsq,
                                int F, int n, int rowsPerBlock) {
    __shared__ float ssum[256];
    __shared__ float ssq[256];
    int tid  = threadIdx.x;
    int chan = tid & (F - 1);
    int grp  = tid / F;
    int step = 256 / F;
    int r0   = blockIdx.x * rowsPerBlock + grp;
    int rEnd = blockIdx.x * rowsPerBlock + rowsPerBlock;
    if (rEnd > n) rEnd = n;
    float b = bias[chan];
    float s = 0.f, q = 0.f;
    for (int r = r0; r < rEnd; r += step) {
        float v = x[(size_t)r * F + chan] + b;
        s += v; q += v * v;
    }
    ssum[tid] = s; ssq[tid] = q;
    __syncthreads();
    if (tid < F) {
        for (int t = tid + F; t < 256; t += F) { s += ssum[t]; q += ssq[t]; }
        atomAddF(&gsum[tid], s);
        atomAddF(&gsq[tid],  q);
    }
}

__global__ void bn_finalize_kernel(const float* __restrict__ gsum, const float* __restrict__ gsq,
                                   const float* __restrict__ gamma, const float* __restrict__ beta,
                                   float* __restrict__ scale, float* __restrict__ shift,
                                   int F, int n) {
    int i = threadIdx.x;
    if (i < F) {
        float inv  = 1.0f / (float)n;
        float mean = gsum[i] * inv;
        float var  = gsq[i] * inv - mean * mean;
        float sc   = gamma[i] * rsqrtf(var + EPSV);
        scale[i] = sc;
        shift[i] = beta[i] - mean * sc;
    }
}

// in-place: x = prelu(bn(x + bias))
__global__ void bn_prelu_kernel(float* __restrict__ x, const float* __restrict__ bias,
                                const float* __restrict__ scale, const float* __restrict__ shift,
                                const float* __restrict__ prelu_a, int F, long long total) {
    long long i = (long long)blockIdx.x * blockDim.x + threadIdx.x;
    if (i >= total) return;
    int ch = (int)(i % F);
    float v = x[i] + bias[ch];
    float y = scale[ch] * v + shift[ch];
    x[i] = (y >= 0.f) ? y : prelu_a[0] * y;
}

// out = bn2(agg2 + b2) + (x@Wr + br)   (residual lives in hcomb cols 64..95)
__global__ void final_kernel(const float* __restrict__ agg2, const float* __restrict__ b2,
                             const float* __restrict__ scale2, const float* __restrict__ shift2,
                             const float* __restrict__ hcomb, const float* __restrict__ br,
                             float* __restrict__ out, long long total) {
    long long i = (long long)blockIdx.x * blockDim.x + threadIdx.x;
    if (i >= total) return;
    int ch = (int)(i % LATDIM);
    long long row = i / LATDIM;
    float v = agg2[i] + b2[ch];
    float y = scale2[ch] * v + shift2[ch];
    out[i] = y + hcomb[row * 96 + HIDDIM + ch] + br[ch];
}

// ---------------- launch ----------------

extern "C" void kernel_launch(void* const* d_in, const int* in_sizes, int n_in,
                              void* d_out, int out_size, void* d_ws, size_t ws_size,
                              hipStream_t stream) {
    const float* x   = (const float*)d_in[0];
    const int*   ei  = (const int*)d_in[1];      // [2 x E] flattened
    const float* ew  = (const float*)d_in[2];
    const float* W1  = (const float*)d_in[3];
    const float* b1  = (const float*)d_in[4];
    const float* g1  = (const float*)d_in[5];
    const float* be1 = (const float*)d_in[6];
    const float* W2  = (const float*)d_in[7];
    const float* b2  = (const float*)d_in[8];
    const float* g2  = (const float*)d_in[9];
    const float* be2 = (const float*)d_in[10];
    const float* pa  = (const float*)d_in[11];
    const float* Wr  = (const float*)d_in[12];
    const float* br  = (const float*)d_in[13];
    float* out = (float*)d_out;

    char* ws = (char*)d_ws;
    size_t off = 0;
    auto carve = [&](size_t bytes) -> char* {
        char* p = ws + off;
        off = (off + bytes + 255) & ~(size_t)255;
        return p;
    };
    float*  dinv   = (float*)carve((size_t)NN * 4);           // deg, then rsqrt in place
    float*  norm   = (float*)carve((size_t)NE * 4);
    __bf16* wc1    = (__bf16*)carve((size_t)INDIM * 96 * 2);  // row-major bf16 [W1|Wr]
    __bf16* wp2    = (__bf16*)carve((size_t)HIDDIM * LATDIM * 2);
    __bf16* wc1pk  = (__bf16*)carve((size_t)INDIM * 96 * 2);  // fragment-major
    __bf16* wp2pk  = (__bf16*)carve((size_t)HIDDIM * LATDIM * 2);
    float*  hcomb  = (float*)carve((size_t)NN * 96 * 4);      // [x@W1 | x@Wr]
    float*  agg1   = (float*)carve((size_t)NN * HIDDIM * 4);  // reused in-place as activations
    float*  h2     = (float*)carve((size_t)NN * LATDIM * 4);
    float*  agg2   = (float*)carve((size_t)NN * LATDIM * 4);
    float*  stats  = (float*)carve(384 * 4);
    float *sum1 = stats,       *sq1 = stats + 64,  *sc1 = stats + 128, *sh1 = stats + 192;
    float *sum2 = stats + 256, *sq2 = stats + 288, *sc2 = stats + 320, *sh2 = stats + 352;

    hipMemsetAsync(dinv,  0, (size_t)NN * 4, stream);
    hipMemsetAsync(agg1,  0, (size_t)NN * HIDDIM * 4, stream);
    hipMemsetAsync(agg2,  0, (size_t)NN * LATDIM * 4, stream);
    hipMemsetAsync(stats, 0, 384 * 4, stream);

    const int B = 256;
    const int nRowTiles = (NN + 15) / 16;   // 6250
    // 1) gcn_norm
    deg_kernel <<<(NE + B - 1) / B, B, 0, stream>>>(ei, ew, dinv, NE);
    dinv_kernel<<<(NN + B - 1) / B, B, 0, stream>>>(dinv, NN);
    norm_kernel<<<(NE + B - 1) / B, B, 0, stream>>>(ei, ew, dinv, norm, NE);

    // 2) pack weights to bf16, then to WMMA fragment-major order
    pack_w1wr<<<(INDIM * 96 + B - 1) / B, B, 0, stream>>>(W1, Wr, wc1);
    pack_w2  <<<(HIDDIM * LATDIM + B - 1) / B, B, 0, stream>>>(W2, wp2);
    reorder_b_kernel<<<(INDIM * 96 + B - 1) / B, B, 0, stream>>>(wc1, 96, INDIM / 32, 6, wc1pk);
    reorder_b_kernel<<<(HIDDIM * LATDIM + B - 1) / B, B, 0, stream>>>(wp2, 32, HIDDIM / 32, 2, wp2pk);

    // 3) fused GEMM: hcomb = x @ [W1 | Wr]; one streaming pass over x, 6 accumulators/wave
    wmma_gemm_bf16<6><<<(nRowTiles + 3) / 4, 128, 0, stream>>>(
        x, INDIM, wc1pk, hcomb, 96, INDIM, nRowTiles);

    // 4) conv1 aggregation: agg1[col] += hcomb[row, 0:64] * norm
    {
        long long tot = (long long)(NE + NN) * (HIDDIM / 2);
        scatter_kernel<<<(int)((tot + B - 1) / B), B, 0, stream>>>(
            hcomb, 96, ei, norm, dinv, agg1, HIDDIM, NE, NN);
    }

    // 5) BN1 + PReLU (bias b1 folded into stats & apply), in place on agg1
    bn_stats_kernel   <<<(NN + 511) / 512, B, 0, stream>>>(agg1, b1, sum1, sq1, HIDDIM, NN, 512);
    bn_finalize_kernel<<<1, 64, 0, stream>>>(sum1, sq1, g1, be1, sc1, sh1, HIDDIM, NN);
    {
        long long tot = (long long)NN * HIDDIM;
        bn_prelu_kernel<<<(int)((tot + B - 1) / B), B, 0, stream>>>(
            agg1, b1, sc1, sh1, pa, HIDDIM, tot);
    }

    // 6) GEMM2: h2 = act @ W2
    wmma_gemm_bf16<2><<<(nRowTiles + 3) / 4, 128, 0, stream>>>(
        agg1, HIDDIM, wp2pk, h2, LATDIM, HIDDIM, nRowTiles);

    // 7) conv2 aggregation
    {
        long long tot = (long long)(NE + NN) * (LATDIM / 2);
        scatter_kernel<<<(int)((tot + B - 1) / B), B, 0, stream>>>(
            h2, LATDIM, ei, norm, dinv, agg2, LATDIM, NE, NN);
    }

    // 8) BN2 + residual add -> out
    bn_stats_kernel   <<<(NN + 511) / 512, B, 0, stream>>>(agg2, b2, sum2, sq2, LATDIM, NN, 512);
    bn_finalize_kernel<<<1, 32, 0, stream>>>(sum2, sq2, g2, be2, sc2, sh2, LATDIM, NN);
    {
        long long tot = (long long)NN * LATDIM;
        final_kernel<<<(int)((tot + B - 1) / B), B, 0, stream>>>(
            agg2, b2, sc2, sh2, hcomb, br, out, tot);
    }
}
